// Tacotron2_5970004542144
// MI455X (gfx1250) — matrix operand order repaired and checked by reading, not compiled, not run on hardware
//
#include <hip/hip_runtime.h>

typedef unsigned short u16;
typedef __attribute__((ext_vector_type(16))) __bf16 v16bf;
typedef __attribute__((ext_vector_type(8)))  float  v8f;

// ---------------------------------------------------------------------------
// dims
#define BB 16
#define LL 160
#define TT 500
#define ENC 512
#define PRE 256
#define ARNN 1024
#define ADIM 128
#define RNNH 1024
#define NMEL 80
#define LOCF 32
#define LOCK 31
#define KA 1792            /* arnn combined K: 256 pre + 512 ctx + 1024 h   */
#define KD 2560            /* drnn combined K: 1024 ah + 512 ctx + 1024 dh  */
#define KP 1536            /* proj K: 1024 dh + 512 ctx                     */
#define DEC_BLOCKS 32      /* 32 blk x 8 waves = 256 waves = N tiles (4096) */

// input index map (setup_inputs dict order, params flattened)
enum {
  IN_CHARS=0, IN_CHARS_LEN=1, IN_MEL=2, IN_MEL_LEN=3,
  P_EMB=4,
  P_EC0=5,                         // per conv i: w,b,g,bb,m,v at 5+6i..10+6i
  P_ELF_WIH=23, P_ELF_WHH=24, P_ELF_BIH=25, P_ELF_BHH=26,
  P_ELB_WIH=27, P_ELB_WHH=28, P_ELB_BIH=29, P_ELB_BHH=30,
  P_PRE_W1=31, P_PRE_W2=32,
  P_ATT_ENC=33, P_ATT_Q=34, P_LOC_CONV=35, P_LOC_LIN=36, P_ATT_V=37,
  P_ARNN_WIH=38, P_ARNN_WHH=39, P_ARNN_BIH=40, P_ARNN_BHH=41,
  P_DRNN_WIH=42, P_DRNN_WHH=43, P_DRNN_BIH=44, P_DRNN_BHH=45,
  P_MEL_W=46, P_MEL_B=47, P_GATE_W=48, P_GATE_B=49,
  P_PC0=50                         // per conv i: w,b,g,bb,m,v at 50+6i..55+6i
};

// ---------------------------------------------------------------------------
// helpers
__device__ __forceinline__ float b2f(u16 u) {
  union { float f; unsigned i; } x; x.i = ((unsigned)u) << 16; return x.f;
}
__device__ __forceinline__ u16 f2b(float f) {
  union { float f; unsigned i; } x; x.f = f;
  unsigned i = x.i;
  return (u16)((i + 0x7fffu + ((i >> 16) & 1u)) >> 16);
}
__device__ __forceinline__ float sigm(float x) { return 1.f / (1.f + __expf(-x)); }
__device__ __forceinline__ v8f zero8() { v8f z; for (int i = 0; i < 8; i++) z[i] = 0.f; return z; }

union FragBF { v16bf v; uint4 q[2]; };

// A fragment, 16x32 bf16 (M x K), A row-major [M, lda], lda mult of 32.
// lanes 0-15: row=lane, K {k..k+15}; lanes 16-31: row=lane-16, K {k+16..k+31}
__device__ __forceinline__ v16bf load_frag_a(const u16* A, int lda, int mBase, int Mmax,
                                             int k, int lane) {
  int row = mBase + (lane & 15);
  if (row >= Mmax) row = Mmax - 1;
  int kOff = (lane & 16) ? 16 : 0;
  const u16* p = A + (size_t)row * lda + k;
  FragBF f;
  f.q[0] = *(const uint4*)(p + kOff);
  f.q[1] = *(const uint4*)(p + 8 + kOff);
  return f.v;
}

// B fragment, 32x16 bf16 (K x N), weights stored row-major [N, ldw] (i.e. W[n][k]).
// lanes 0-15: col=lane, K {k..k+15}; lanes 16-31: col=lane-16, K {k+16..k+31}
__device__ __forceinline__ v16bf load_frag_b(const u16* W, int ldw, int nBase,
                                             int k, int lane) {
  int col = nBase + (lane & 15);
  int kOff = (lane & 16) ? 16 : 0;
  const u16* p = W + (size_t)col * ldw + k + kOff;
  FragBF f;
  f.q[0] = ((const uint4*)p)[0];
  f.q[1] = ((const uint4*)p)[1];
  return f.v;
}

// C/D 16x16 f32: VGPR r -> row r (lanes 0-15) / row r+8 (lanes 16-31), col = lane&15
__device__ __forceinline__ void store_tile(float* C, int ldc, int mBase, int Mmax,
                                           int nBase, int Nmax, v8f acc, int lane) {
  int col = nBase + (lane & 15);
  int r0 = (lane & 16) ? 8 : 0;
  if (col >= Nmax) return;
  for (int r = 0; r < 8; r++) {
    int row = mBase + r0 + r;
    if (row < Mmax) C[(size_t)row * ldc + col] = acc[r];
  }
}

__device__ __forceinline__ v8f wmma_bf16(v16bf a, v16bf b, v8f c) {
  return __builtin_amdgcn_wmma_f32_16x16x32_bf16(false, a, false, b, (short)0, c, false, false);
}

// grid-wide generation barrier (bar[0]=arrive count, bar[1]=generation)
__device__ void gbar(int* bar, int nb) {
  __syncthreads();
  __threadfence();
  if (threadIdx.x == 0) {
    int gen = atomicAdd(bar + 1, 0);
    if (atomicAdd(bar, 1) == nb - 1) {
      atomicExch(bar, 0);
      __threadfence();
      atomicAdd(bar + 1, 1);
    } else {
      while (atomicAdd(bar + 1, 0) == gen) __builtin_amdgcn_s_sleep(2);
    }
  }
  __syncthreads();
}

// ---------------------------------------------------------------------------
// generic bf16 GEMM: C[M,N] = A[M,K] * W[N,K]^T, K mult of 32, N mult of 16
__global__ void k_gemm(const u16* __restrict__ A, const u16* __restrict__ W,
                       float* __restrict__ C, int M, int N, int K) {
  int lane = threadIdx.x & 31;
  int gw = blockIdx.x * (blockDim.x >> 5) + (threadIdx.x >> 5);
  int nw = gridDim.x * (blockDim.x >> 5);
  int mT = (M + 15) >> 4, nT = N >> 4;
  long total = (long)mT * nT;
  for (long tile = gw; tile < total; tile += nw) {
    int mt = (int)(tile / nT), nt = (int)(tile - (long)mt * nT);
    v8f acc = zero8();
    const u16* wrow = W + (size_t)(nt * 16 + (lane & 15)) * K;
    for (int k = 0; k < K; k += 32) {
      __builtin_prefetch((const void*)(wrow + k + 256), 0, 1);   // global_prefetch_b8
      v16bf a = load_frag_a(A, K, mt * 16, M, k, lane);
      v16bf b = load_frag_b(W, K, nt * 16, k, lane);
      acc = wmma_bf16(a, b, acc);
    }
    store_tile(C, N, mt * 16, M, nt * 16, N, acc, lane);
  }
}

// ---------------------------------------------------------------------------
// small elementwise / packing kernels
__global__ void k_cvt_pad(const float* __restrict__ s, u16* __restrict__ d,
                          long rows, int scols, int dcols) {
  long n = rows * dcols;
  for (long i = blockIdx.x * (long)blockDim.x + threadIdx.x; i < n; i += (long)blockDim.x * gridDim.x) {
    long r = i / dcols; int c = (int)(i - r * dcols);
    d[i] = f2b(c < scols ? s[r * scols + c] : 0.f);
  }
}

__global__ void k_concat2(const float* __restrict__ s1, int K1,
                          const float* __restrict__ s2, int K2,
                          u16* __restrict__ d, int N) {
  int K = K1 + K2;
  long n = (long)N * K;
  for (long i = blockIdx.x * (long)blockDim.x + threadIdx.x; i < n; i += (long)blockDim.x * gridDim.x) {
    long r = i / K; int k = (int)(i - r * K);
    float v = (k < K1) ? s1[r * K1 + k] : s2[r * K2 + (k - K1)];
    d[i] = f2b(v);
  }
}

__global__ void k_projpack(const float* __restrict__ melw, const float* __restrict__ gatew,
                           u16* __restrict__ d) {
  long n = 96L * KP;
  for (long i = blockIdx.x * (long)blockDim.x + threadIdx.x; i < n; i += (long)blockDim.x * gridDim.x) {
    int r = (int)(i / KP), k = (int)(i % KP);
    float v = (r < NMEL) ? melw[(long)r * KP + k] : (r == NMEL ? gatew[k] : 0.f);
    d[i] = f2b(v);
  }
}

__global__ void k_addvec(const float* a, const float* b, float* d, int n) {
  for (int i = blockIdx.x * blockDim.x + threadIdx.x; i < n; i += blockDim.x * gridDim.x)
    d[i] = a[i] + b[i];
}

__global__ void k_embed(const int* __restrict__ chars, const float* __restrict__ emb,
                        float* __restrict__ out) {
  long n = (long)BB * LL * ENC;
  for (long i = blockIdx.x * (long)blockDim.x + threadIdx.x; i < n; i += (long)blockDim.x * gridDim.x) {
    long m = i / ENC; int c = (int)(i % ENC);
    out[i] = emb[(long)chars[m] * ENC + c];
  }
}

// im2col for K=5, pad=2 conv over positions (rows m = b*Lt + pos, src [m, C] f32)
__global__ void k_im2col(const float* __restrict__ x, u16* __restrict__ d,
                         int Lt, int C, int Kp, long rows) {
  long n = rows * Kp;
  for (long i = blockIdx.x * (long)blockDim.x + threadIdx.x; i < n; i += (long)blockDim.x * gridDim.x) {
    long m = i / Kp; int k = (int)(i - m * Kp);
    float v = 0.f;
    if (k < C * 5) {
      int ci = k / 5, j = k % 5;
      int pos = (int)(m % Lt); long b = m / Lt;
      int ps = pos - 2 + j;
      if (ps >= 0 && ps < Lt) v = x[(b * Lt + ps) * (long)C + ci];
    }
    d[i] = f2b(v);
  }
}

__global__ void k_bnact(const float* __restrict__ x, float* __restrict__ out, long n, int C,
                        const float* cb, const float* g, const float* bb,
                        const float* mm, const float* vv, int act) {
  for (long i = blockIdx.x * (long)blockDim.x + threadIdx.x; i < n; i += (long)blockDim.x * gridDim.x) {
    int c = (int)(i % C);
    float y = (x[i] + cb[c] - mm[c]) * (g[c] * rsqrtf(vv[c] + 1e-5f)) + bb[c];
    out[i] = (act == 0) ? fmaxf(y, 0.f) : tanhf(y);
  }
}

__global__ void k_relu_cvt(const float* __restrict__ s, u16* __restrict__ d, long n) {
  for (long i = blockIdx.x * (long)blockDim.x + threadIdx.x; i < n; i += (long)blockDim.x * gridDim.x)
    d[i] = f2b(fmaxf(s[i], 0.f));
}

// dec_in[b,t] = (t>0) ? mel[b,t-1] : 0, padded to 96 cols, bf16
__global__ void k_decin(const float* __restrict__ mel, u16* __restrict__ d) {
  long n = (long)BB * TT * 96;
  for (long i = blockIdx.x * (long)blockDim.x + threadIdx.x; i < n; i += (long)blockDim.x * gridDim.x) {
    long m = i / 96; int c = (int)(i % 96);
    int t = (int)(m % TT); long b = m / TT;
    float v = (c < NMEL && t > 0) ? mel[(b * TT + (t - 1)) * NMEL + c] : 0.f;
    d[i] = f2b(v);
  }
}

// ---------------------------------------------------------------------------
// BiLSTM scan: 2 blocks (dir), h/c in LDS, gates = pre[l] + h@Whh^T via WMMA (A from LDS)
__global__ void k_bilstm(const float* __restrict__ preF, const float* __restrict__ preB,
                         const u16* __restrict__ whhF, const u16* __restrict__ whhB,
                         const float* __restrict__ bsF, const float* __restrict__ bsB,
                         float* __restrict__ gscr, u16* __restrict__ encoded) {
  const int dir = blockIdx.x;
  const float* pre  = dir ? preB : preF;
  const u16*   whh  = dir ? whhB : whhF;
  const float* bsum = dir ? bsB : bsF;
  float* gsc = gscr + (size_t)dir * BB * 1024;
  __shared__ u16  s_h[BB * 256];
  __shared__ float s_c[BB * 256];
  for (int e = threadIdx.x; e < BB * 256; e += blockDim.x) { s_h[e] = 0; s_c[e] = 0.f; }
  __syncthreads();
  const int lane = threadIdx.x & 31, wv = threadIdx.x >> 5;
  for (int step = 0; step < LL; step++) {
    int l = dir ? (LL - 1 - step) : step;
    for (int nt = wv; nt < 64; nt += 8) {               // N = 1024 gates
      v8f acc = zero8();
      for (int k = 0; k < 256; k += 32) {
        v16bf a = load_frag_a((const u16*)s_h, 256, 0, BB, k, lane);  // ds_load_b128
        v16bf b = load_frag_b(whh, 256, nt * 16, k, lane);
        acc = wmma_bf16(a, b, acc);
      }
      store_tile(gsc, 1024, 0, BB, nt * 16, 1024, acc, lane);
    }
    __syncthreads();
    for (int e = threadIdx.x; e < BB * 256; e += blockDim.x) {
      int b = e >> 8, j = e & 255;
      const float* gr = gsc + (size_t)b * 1024;
      const float* pr = pre + ((size_t)b * LL + l) * 1024;
      float gi = gr[j]       + pr[j]       + bsum[j];
      float gf = gr[256 + j] + pr[256 + j] + bsum[256 + j];
      float gg = gr[512 + j] + pr[512 + j] + bsum[512 + j];
      float go = gr[768 + j] + pr[768 + j] + bsum[768 + j];
      float c = sigm(gf) * s_c[e] + sigm(gi) * tanhf(gg);
      float h = sigm(go) * tanhf(c);
      s_c[e] = c; s_h[e] = f2b(h);
      encoded[((size_t)b * LL + l) * ENC + dir * 256 + j] = f2b(h);
    }
    __syncthreads();
  }
}

// ---------------------------------------------------------------------------
// persistent decoder
struct DecArgs {
  const u16 *Wa, *Wd, *Wp;           // bf16 [4096,KA], [4096,KD], [96,KP]
  const u16 *prenet_out;             // bf16 [B*T, 256]
  const u16 *encoded;                // bf16 [B*L, 512]
  const float *attenc;               // f32  [B*L, 128]
  const float *attq;                 // f32  [128,1024]
  const float *locconv;              // f32  [32,2,31]
  const float *loclin;               // f32  [128,32]
  const float *attv;                 // f32  [128]
  const float *bsum_a, *bsum_d;      // f32  [4096]
  const float *mel_b, *gate_bp;
  const int *chars_len, *mel_len;
  u16 *xa, *xd, *xp;                 // bf16 [16,KA],[16,KD],[16,KP]
  float *gatesA, *gatesD;            // f32  [16,4096]
  float *ah, *ac, *dh, *dc;          // f32  [16,1024]
  float *aw, *awc;                   // f32  [16,160]
  float *out_mels, *out_gates, *out_align;
  int *bar;
};

__device__ void attention_step(const DecArgs& A, int b, int t) {
  __shared__ float s_q[ADIM];
  __shared__ float s_e[LL];
  __shared__ float s_inv[1];
  int tx = threadIdx.x;
  if (tx < ADIM) {                                   // q = ah[b] @ attq^T
    float s = 0.f;
    const float* hr = A.ah + (size_t)b * ARNN;
    const float* wr = A.attq + (size_t)tx * ARNN;
    for (int k = 0; k < ARNN; k++) s += hr[k] * wr[k];
    s_q[tx] = s;
  }
  __syncthreads();
  if (tx < LL) {                                     // location conv + energy
    float lc[LOCF];
    const float* awr  = A.aw  + (size_t)b * LL;
    const float* awcr = A.awc + (size_t)b * LL;
    for (int f = 0; f < LOCF; f++) {
      float s = 0.f;
      const float* w0 = A.locconv + (size_t)(f * 2 + 0) * LOCK;
      const float* w1 = A.locconv + (size_t)(f * 2 + 1) * LOCK;
      for (int j = 0; j < LOCK; j++) {
        int l = tx - 15 + j;
        if (l >= 0 && l < LL) s += w0[j] * awr[l] + w1[j] * awcr[l];
      }
      lc[f] = s;
    }
    const float* aer = A.attenc + ((size_t)b * LL + tx) * ADIM;
    float e = 0.f;
    for (int a = 0; a < ADIM; a++) {
      float loc = 0.f;
      const float* lw = A.loclin + (size_t)a * LOCF;
      for (int f = 0; f < LOCF; f++) loc += lw[f] * lc[f];
      e += A.attv[a] * tanhf(s_q[a] + aer[a] + loc);
    }
    if (tx >= A.chars_len[b]) e = -1e9f;
    s_e[tx] = e;
  }
  __syncthreads();
  if (tx == 0) {                                     // softmax (160 elems)
    float mx = -1e30f;
    for (int l = 0; l < LL; l++) mx = fmaxf(mx, s_e[l]);
    float s = 0.f;
    for (int l = 0; l < LL; l++) { float p = __expf(s_e[l] - mx); s_e[l] = p; s += p; }
    s_inv[0] = 1.f / s;
  }
  __syncthreads();
  float inv = s_inv[0];
  if (tx < LL) {
    float p = s_e[tx] * inv;
    s_e[tx] = p;
    A.aw[(size_t)b * LL + tx] = p;
    A.awc[(size_t)b * LL + tx] += p;
    A.out_align[((size_t)b * TT + t) * LL + tx] = p;
  }
  __syncthreads();
  for (int d = tx; d < ENC; d += blockDim.x) {       // ctx = aw . encoded
    float s = 0.f;
    const u16* er = A.encoded + (size_t)b * LL * ENC + d;
    for (int l = 0; l < LL; l++) s += s_e[l] * b2f(er[(size_t)l * ENC]);
    u16 cb = f2b(s);
    A.xa[(size_t)b * KA + 256 + d] = cb;
    A.xd[(size_t)b * KD + 1024 + d] = cb;
    A.xp[(size_t)b * KP + 1024 + d] = cb;
  }
}

__global__ void k_decoder(DecArgs A) {
  const int lane = threadIdx.x & 31;
  const int gw = blockIdx.x * (blockDim.x >> 5) + (threadIdx.x >> 5);   // 0..255
  const int tid = blockIdx.x * blockDim.x + threadIdx.x;
  const int NT = blockDim.x * gridDim.x;
  for (int t = 0; t < TT; t++) {
    // ---- Phase A: gatesA = xa @ Wa^T  (M=16, N=4096, K=1792)
    {
      v8f acc = zero8();
      const u16* wrow = A.Wa + (size_t)(gw * 16 + (lane & 15)) * KA;
      for (int k = 0; k < KA; k += 32) {
        __builtin_prefetch((const void*)(wrow + k + 256), 0, 1);
        v16bf a = load_frag_a(A.xa, KA, 0, BB, k, lane);
        v16bf b = load_frag_b(A.Wa, KA, gw * 16, k, lane);
        acc = wmma_bf16(a, b, acc);
      }
      store_tile(A.gatesA, 4096, 0, BB, gw * 16, 4096, acc, lane);
    }
    gbar(A.bar, gridDim.x);
    // ---- Phase B: arnn LSTM pointwise -> ah, ac
    for (int e = tid; e < BB * ARNN; e += NT) {
      int b = e >> 10, j = e & 1023;
      const float* g = A.gatesA + (size_t)b * 4096;
      float gi = g[j]        + A.bsum_a[j];
      float gf = g[1024 + j] + A.bsum_a[1024 + j];
      float gg = g[2048 + j] + A.bsum_a[2048 + j];
      float go = g[3072 + j] + A.bsum_a[3072 + j];
      float c = sigm(gf) * A.ac[e] + sigm(gi) * tanhf(gg);
      float h = sigm(go) * tanhf(c);
      A.ac[e] = c; A.ah[e] = h;
      u16 hb = f2b(h);
      A.xa[(size_t)b * KA + 768 + j] = hb;
      A.xd[(size_t)b * KD + j] = hb;
    }
    gbar(A.bar, gridDim.x);
    // ---- Phase C: attention (block b handles batch b)
    if (blockIdx.x < BB) attention_step(A, blockIdx.x, t);
    gbar(A.bar, gridDim.x);
    // ---- Phase D: gatesD = xd @ Wd^T  (M=16, N=4096, K=2560)
    {
      v8f acc = zero8();
      const u16* wrow = A.Wd + (size_t)(gw * 16 + (lane & 15)) * KD;
      for (int k = 0; k < KD; k += 32) {
        __builtin_prefetch((const void*)(wrow + k + 256), 0, 1);
        v16bf a = load_frag_a(A.xd, KD, 0, BB, k, lane);
        v16bf b = load_frag_b(A.Wd, KD, gw * 16, k, lane);
        acc = wmma_bf16(a, b, acc);
      }
      store_tile(A.gatesD, 4096, 0, BB, gw * 16, 4096, acc, lane);
    }
    gbar(A.bar, gridDim.x);
    // ---- Phase E: drnn LSTM pointwise -> dh, dc
    for (int e = tid; e < BB * RNNH; e += NT) {
      int b = e >> 10, j = e & 1023;
      const float* g = A.gatesD + (size_t)b * 4096;
      float gi = g[j]        + A.bsum_d[j];
      float gf = g[1024 + j] + A.bsum_d[1024 + j];
      float gg = g[2048 + j] + A.bsum_d[2048 + j];
      float go = g[3072 + j] + A.bsum_d[3072 + j];
      float c = sigm(gf) * A.dc[e] + sigm(gi) * tanhf(gg);
      float h = sigm(go) * tanhf(c);
      A.dc[e] = c; A.dh[e] = h;
      u16 hb = f2b(h);
      A.xd[(size_t)b * KD + 1536 + j] = hb;
      A.xp[(size_t)b * KP + j] = hb;
    }
    gbar(A.bar, gridDim.x);
    // ---- Phase F: projection [mel|gate] = xp @ Wp^T + stage next prenet row
    if (gw < 6) {
      v8f acc = zero8();
      for (int k = 0; k < KP; k += 32) {
        v16bf a = load_frag_a(A.xp, KP, 0, BB, k, lane);
        v16bf b = load_frag_b(A.Wp, KP, gw * 16, k, lane);
        acc = wmma_bf16(a, b, acc);
      }
      int n = gw * 16 + (lane & 15);
      int r0 = (lane & 16) ? 8 : 0;
      for (int r = 0; r < 8; r++) {
        int b = r0 + r;
        float val = acc[r];
        if (n < NMEL)       A.out_mels[(size_t)b * TT * NMEL + (size_t)t * NMEL + n] = val + A.mel_b[n];
        else if (n == NMEL) A.out_gates[(size_t)b * TT + t] = val + A.gate_bp[0];
      }
    }
    if (blockIdx.x == gridDim.x - 1 && t + 1 < TT) {
      for (int e = threadIdx.x; e < BB * PRE; e += blockDim.x) {
        int b = e >> 8, j = e & 255;
        A.xa[(size_t)b * KA + j] = A.prenet_out[((size_t)b * TT + t + 1) * PRE + j];
      }
    }
    gbar(A.bar, gridDim.x);
  }
}

// final: bn on last postnet conv, residual, masking of mels/mels_post/gates
__global__ void k_final(const float* __restrict__ y, const float* cb, const float* g,
                        const float* bb, const float* mm, const float* vv,
                        float* mels, float* melsp, float* gates, const int* mlen) {
  long n = (long)BB * TT * NMEL;
  for (long i = blockIdx.x * (long)blockDim.x + threadIdx.x; i < n; i += (long)blockDim.x * gridDim.x) {
    long m = i / NMEL; int c = (int)(i % NMEL);
    int t = (int)(m % TT); long b = m / TT;
    float bn = (y[i] + cb[c] - mm[c]) * (g[c] * rsqrtf(vv[c] + 1e-5f)) + bb[c];
    bool keep = t < mlen[b];
    float mv = mels[i];
    melsp[i] = keep ? (mv + bn) : 0.f;
    mels[i]  = keep ? mv : 0.f;
    if (c == 0) { float gv = gates[m]; gates[m] = keep ? gv : -1000.f; }
  }
}

// ---------------------------------------------------------------------------
#define GS(n) ((int)((((long)(n)) + 255) / 256) < 4096 ? (int)((((long)(n)) + 255) / 256) : 4096)

extern "C" void kernel_launch(void* const* d_in, const int* in_sizes, int n_in,
                              void* d_out, int out_size, void* d_ws, size_t ws_size,
                              hipStream_t stream) {
  (void)in_sizes; (void)n_in; (void)out_size; (void)ws_size;
  const int*   chars  = (const int*)d_in[IN_CHARS];
  const int*   clen   = (const int*)d_in[IN_CHARS_LEN];
  const float* mel_in = (const float*)d_in[IN_MEL];
  const int*   mlen   = (const int*)d_in[IN_MEL_LEN];
  const float* F(int i) ; // fwd decl trick not allowed; use lambda-free macro:
#define PF(i) ((const float*)d_in[(i)])

  // ---- workspace bump allocator (256B aligned)
  char* base = (char*)d_ws; size_t off = 0;
  auto ALLOC = [&](size_t bytes) -> void* {
    void* p = base + off; off = (off + bytes + 255) & ~(size_t)255; return p;
  };
  // bf16 weights
  u16* Wa    = (u16*)ALLOC((size_t)4096 * KA * 2);
  u16* Wd    = (u16*)ALLOC((size_t)4096 * KD * 2);
  u16* Wp    = (u16*)ALLOC((size_t)96 * KP * 2);
  u16* ecw[3]; for (int i = 0; i < 3; i++) ecw[i] = (u16*)ALLOC((size_t)ENC * 2560 * 2);
  u16* elwihF = (u16*)ALLOC((size_t)1024 * ENC * 2);
  u16* elwihB = (u16*)ALLOC((size_t)1024 * ENC * 2);
  u16* elwhhF = (u16*)ALLOC((size_t)1024 * 256 * 2);
  u16* elwhhB = (u16*)ALLOC((size_t)1024 * 256 * 2);
  u16* prew1  = (u16*)ALLOC((size_t)PRE * 96 * 2);
  u16* prew2  = (u16*)ALLOC((size_t)PRE * PRE * 2);
  u16* attew  = (u16*)ALLOC((size_t)ADIM * ENC * 2);
  const int pch[6] = { NMEL, 512, 512, 512, 512, NMEL };
  const int pkp[5] = { 416, 2560, 2560, 2560, 2560 };
  u16* pcw[5]; for (int i = 0; i < 5; i++) pcw[i] = (u16*)ALLOC((size_t)pch[i+1] * pkp[i] * 2);
  float* bsum_a = (float*)ALLOC(4096 * 4);
  float* bsum_d = (float*)ALLOC(4096 * 4);
  float* bsLF   = (float*)ALLOC(1024 * 4);
  float* bsLB   = (float*)ALLOC(1024 * 4);
  // activations
  float* F1   = (float*)ALLOC((size_t)8000 * 512 * 4);
  float* F2   = (float*)ALLOC((size_t)8000 * 512 * 4);
  u16*  colB  = (u16*)ALLOC((size_t)8000 * 2560 * 2);
  float* preF = (float*)ALLOC((size_t)2560 * 1024 * 4);
  float* preB = (float*)ALLOC((size_t)2560 * 1024 * 4);
  u16*  enc   = (u16*)ALLOC((size_t)2560 * ENC * 2);
  float* atte = (float*)ALLOC((size_t)2560 * ADIM * 4);
  u16*  pnet  = (u16*)ALLOC((size_t)8000 * PRE * 2);
  float* lstmg = (float*)ALLOC((size_t)2 * BB * 1024 * 4);
  // decoder state
  u16*  xa = (u16*)ALLOC((size_t)BB * KA * 2);
  u16*  xd = (u16*)ALLOC((size_t)BB * KD * 2);
  u16*  xp = (u16*)ALLOC((size_t)BB * KP * 2);
  float* gatesA = (float*)ALLOC((size_t)BB * 4096 * 4);
  float* gatesD = (float*)ALLOC((size_t)BB * 4096 * 4);
  float* ah = (float*)ALLOC((size_t)BB * 1024 * 4);
  float* ac = (float*)ALLOC((size_t)BB * 1024 * 4);
  float* dh = (float*)ALLOC((size_t)BB * 1024 * 4);
  float* dc = (float*)ALLOC((size_t)BB * 1024 * 4);
  float* aw = (float*)ALLOC((size_t)BB * LL * 4);
  float* awc = (float*)ALLOC((size_t)BB * LL * 4);
  int*  bar = (int*)ALLOC(256);

  float* out_mels  = (float*)d_out;
  float* out_post  = out_mels + (size_t)BB * TT * NMEL;
  float* out_gates = out_post + (size_t)BB * TT * NMEL;
  float* out_align = out_gates + (size_t)BB * TT;

  // ---- zero the decoder state (graph-capture safe, deterministic per launch)
  hipMemsetAsync(xa, 0, (size_t)BB * KA * 2, stream);
  hipMemsetAsync(xd, 0, (size_t)BB * KD * 2, stream);
  hipMemsetAsync(xp, 0, (size_t)BB * KP * 2, stream);
  hipMemsetAsync(ah, 0, (size_t)BB * 1024 * 4, stream);
  hipMemsetAsync(ac, 0, (size_t)BB * 1024 * 4, stream);
  hipMemsetAsync(dh, 0, (size_t)BB * 1024 * 4, stream);
  hipMemsetAsync(dc, 0, (size_t)BB * 1024 * 4, stream);
  hipMemsetAsync(aw, 0, (size_t)BB * LL * 4, stream);
  hipMemsetAsync(awc, 0, (size_t)BB * LL * 4, stream);
  hipMemsetAsync(bar, 0, 256, stream);

  // ---- weight prep (fp32 -> bf16, concat/pad)
  k_concat2<<<GS(4096L*KA),256,0,stream>>>(PF(P_ARNN_WIH), 768, PF(P_ARNN_WHH), 1024, Wa, 4096);
  k_concat2<<<GS(4096L*KD),256,0,stream>>>(PF(P_DRNN_WIH), 1536, PF(P_DRNN_WHH), 1024, Wd, 4096);
  k_projpack<<<GS(96L*KP),256,0,stream>>>(PF(P_MEL_W), PF(P_GATE_W), Wp);
  for (int i = 0; i < 3; i++)
    k_cvt_pad<<<GS(512L*2560),256,0,stream>>>(PF(P_EC0 + 6*i), ecw[i], 512, 2560, 2560);
  k_cvt_pad<<<GS(1024L*512),256,0,stream>>>(PF(P_ELF_WIH), elwihF, 1024, 512, 512);
  k_cvt_pad<<<GS(1024L*512),256,0,stream>>>(PF(P_ELB_WIH), elwihB, 1024, 512, 512);
  k_cvt_pad<<<GS(1024L*256),256,0,stream>>>(PF(P_ELF_WHH), elwhhF, 1024, 256, 256);
  k_cvt_pad<<<GS(1024L*256),256,0,stream>>>(PF(P_ELB_WHH), elwhhB, 1024, 256, 256);
  k_cvt_pad<<<GS(256L*96),256,0,stream>>>(PF(P_PRE_W1), prew1, 256, 80, 96);
  k_cvt_pad<<<GS(256L*256),256,0,stream>>>(PF(P_PRE_W2), prew2, 256, 256, 256);
  k_cvt_pad<<<GS(128L*512),256,0,stream>>>(PF(P_ATT_ENC), attew, 128, 512, 512);
  for (int i = 0; i < 5; i++)
    k_cvt_pad<<<GS((long)pch[i+1]*pkp[i]),256,0,stream>>>(PF(P_PC0 + 6*i), pcw[i],
                                                          pch[i+1], pch[i]*5, pkp[i]);
  k_addvec<<<GS(4096),256,0,stream>>>(PF(P_ARNN_BIH), PF(P_ARNN_BHH), bsum_a, 4096);
  k_addvec<<<GS(4096),256,0,stream>>>(PF(P_DRNN_BIH), PF(P_DRNN_BHH), bsum_d, 4096);
  k_addvec<<<GS(1024),256,0,stream>>>(PF(P_ELF_BIH), PF(P_ELF_BHH), bsLF, 1024);
  k_addvec<<<GS(1024),256,0,stream>>>(PF(P_ELB_BIH), PF(P_ELB_BHH), bsLB, 1024);

  // ---- encoder: embedding -> 3x(conv+BN+relu) via im2col WMMA GEMM
  k_embed<<<GS(2560L*ENC),256,0,stream>>>(chars, PF(P_EMB), F2);
  for (int i = 0; i < 3; i++) {
    k_im2col<<<GS(2560L*2560),256,0,stream>>>(F2, colB, LL, ENC, 2560, 2560);
    k_gemm<<<256,256,0,stream>>>(colB, ecw[i], F1, 2560, ENC, 2560);
    k_bnact<<<GS(2560L*ENC),256,0,stream>>>(F1, F2, 2560L*ENC, ENC,
        PF(P_EC0+6*i+1), PF(P_EC0+6*i+2), PF(P_EC0+6*i+3), PF(P_EC0+6*i+4), PF(P_EC0+6*i+5), 0);
  }
  // BiLSTM: hoist x@Wih to one GEMM per direction, then LDS-resident scan
  k_cvt_pad<<<GS(2560L*ENC),256,0,stream>>>(F2, colB, 2560, ENC, ENC);
  k_gemm<<<256,256,0,stream>>>(colB, elwihF, preF, 2560, 1024, ENC);
  k_gemm<<<256,256,0,stream>>>(colB, elwihB, preB, 2560, 1024, ENC);
  k_bilstm<<<2,256,0,stream>>>(preF, preB, elwhhF, elwhhB, bsLF, bsLB, lstmg, enc);
  k_gemm<<<64,256,0,stream>>>(enc, attew, atte, 2560, ADIM, ENC);

  // ---- prenet over teacher-forced inputs (all T at once)
  k_decin<<<GS(8000L*96),256,0,stream>>>(mel_in, colB);
  k_gemm<<<256,256,0,stream>>>(colB, prew1, F1, 8000, PRE, 96);
  k_relu_cvt<<<GS(8000L*PRE),256,0,stream>>>(F1, colB, 8000L*PRE);
  k_gemm<<<256,256,0,stream>>>(colB, prew2, F1, 8000, PRE, PRE);
  k_relu_cvt<<<GS(8000L*PRE),256,0,stream>>>(F1, pnet, 8000L*PRE);

  // ---- persistent autoregressive decoder (500 steps, grid-wide barriers)
  DecArgs DA;
  DA.Wa = Wa; DA.Wd = Wd; DA.Wp = Wp;
  DA.prenet_out = pnet; DA.encoded = enc; DA.attenc = atte;
  DA.attq = PF(P_ATT_Q); DA.locconv = PF(P_LOC_CONV); DA.loclin = PF(P_LOC_LIN);
  DA.attv = PF(P_ATT_V); DA.bsum_a = bsum_a; DA.bsum_d = bsum_d;
  DA.mel_b = PF(P_MEL_B); DA.gate_bp = PF(P_GATE_B);
  DA.chars_len = clen; DA.mel_len = mlen;
  DA.xa = xa; DA.xd = xd; DA.xp = xp;
  DA.gatesA = gatesA; DA.gatesD = gatesD;
  DA.ah = ah; DA.ac = ac; DA.dh = dh; DA.dc = dc;
  DA.aw = aw; DA.awc = awc;
  DA.out_mels = out_mels; DA.out_gates = out_gates; DA.out_align = out_align;
  DA.bar = bar;
  k_decoder<<<DEC_BLOCKS,256,0,stream>>>(DA);

  // ---- postnet: 5x(conv+BN[+tanh]) via im2col WMMA GEMM, residual + masks
  const float* psrc = out_mels;
  for (int i = 0; i < 5; i++) {
    k_im2col<<<GS(8000L*pkp[i]),256,0,stream>>>(psrc, colB, TT, pch[i], pkp[i], 8000);
    k_gemm<<<256,256,0,stream>>>(colB, pcw[i], F1, 8000, pch[i+1], pkp[i]);
    if (i < 4) {
      k_bnact<<<GS(8000L*pch[i+1]),256,0,stream>>>(F1, F2, 8000L*pch[i+1], pch[i+1],
          PF(P_PC0+6*i+1), PF(P_PC0+6*i+2), PF(P_PC0+6*i+3), PF(P_PC0+6*i+4), PF(P_PC0+6*i+5), 1);
      psrc = F2;
    }
  }
  k_final<<<GS(8000L*NMEL),256,0,stream>>>(F1,
      PF(P_PC0+6*4+1), PF(P_PC0+6*4+2), PF(P_PC0+6*4+3), PF(P_PC0+6*4+4), PF(P_PC0+6*4+5),
      out_mels, out_post, out_gates, mlen);
#undef PF
}
